// LCNN_79688823210661
// MI455X (gfx1250) — matrix operand order repaired and checked by reading, not compiled, not run on hardware
//
#include <hip/hip_runtime.h>
#include <math.h>

#define L_      8
#define D_      4
#define NSITE   8192          // B * L^4 = 2*4096
#define NVAR_   21
#define NOUT_   8
#define NCH_    8
#define KDIM    252           // NVAR * D * NK
#define KPAD    256
#define KSTEPS  (KPAD / 32)   // 8 WMMA K-steps
#define SPW     16            // sites per workgroup in the WMMA kernel
#define MROWS   (SPW * 18)    // 288 rows: 18 = 9 complex entries * {re,im}
#define ASTR    264           // f16 row stride (252 data + pad, keeps b128 loads legal)
#define WGT     128           // 4 wave32s

typedef __attribute__((ext_vector_type(16))) _Float16 v16h;
typedef __attribute__((ext_vector_type(8)))  float    v8f;

// ---------------- complex 3x3 helpers ----------------
struct C3 { float re[9]; float im[9]; };

__device__ inline int shift_site(int s, int mu, int d) {
  // s = b*4096 + x0*512 + x1*64 + x2*8 + x3 ; periodic shift of x_mu by d
  int b = s >> 12, r = s & 4095;
  int sh = 3 * (3 - mu);
  int x = (((r >> sh) & 7) + d) & 7;
  r = (r & ~(7 << sh)) | (x << sh);
  return (b << 12) | r;
}

__device__ inline C3 cmul(const C3& A, const C3& B) { // A @ B
  C3 C;
#pragma unroll
  for (int a = 0; a < 3; ++a)
#pragma unroll
    for (int b = 0; b < 3; ++b) {
      float r = 0.f, i = 0.f;
#pragma unroll
      for (int k = 0; k < 3; ++k) {
        float ar = A.re[a*3+k], ai = A.im[a*3+k];
        float br = B.re[k*3+b], bi = B.im[k*3+b];
        r = fmaf(ar, br, fmaf(-ai, bi, r));
        i = fmaf(ar, bi, fmaf( ai, br, i));
      }
      C.re[a*3+b] = r; C.im[a*3+b] = i;
    }
  return C;
}

__device__ inline C3 cmul_dagB(const C3& A, const C3& B) { // A @ B^dagger
  C3 C;
#pragma unroll
  for (int a = 0; a < 3; ++a)
#pragma unroll
    for (int b = 0; b < 3; ++b) {
      float r = 0.f, i = 0.f;
#pragma unroll
      for (int k = 0; k < 3; ++k) {
        float ar = A.re[a*3+k], ai = A.im[a*3+k];
        float br = B.re[b*3+k], bi = B.im[b*3+k]; // conj(B[b,k])
        r = fmaf(ar, br, fmaf( ai, bi, r));
        i = fmaf(ai, br, fmaf(-ar, bi, i));
      }
      C.re[a*3+b] = r; C.im[a*3+b] = i;
    }
  return C;
}

__device__ inline C3 cdagger(const C3& A) {
  C3 C;
#pragma unroll
  for (int a = 0; a < 3; ++a)
#pragma unroll
    for (int b = 0; b < 3; ++b) { C.re[a*3+b] = A.re[b*3+a]; C.im[a*3+b] = -A.im[b*3+a]; }
  return C;
}

// ---------------- loaders / stores ----------------
struct SplitLoader { // U_re / U_im planar float arrays: [site][mu][3][3]
  const float* re; const float* im;
  __device__ C3 operator()(int s, int mu) const {
    C3 M; int base = (s * 4 + mu) * 9;
#pragma unroll
    for (int e = 0; e < 9; ++e) { M.re[e] = re[base + e]; M.im[e] = im[base + e]; }
    return M;
  }
};
struct InterLoader { // interleaved complex: [site][mu][3][3][{re,im}]
  const float* p;
  __device__ C3 operator()(int s, int mu) const {
    C3 M; int base = (s * 4 + mu) * 18;
#pragma unroll
    for (int e = 0; e < 9; ++e) { M.re[e] = p[base + 2*e]; M.im[e] = p[base + 2*e + 1]; }
    return M;
  }
};

__device__ inline C3 loadW(const float* W, int s, int ch) {
  C3 M; int base = (s * NVAR_ + ch) * 18;
#pragma unroll
  for (int e = 0; e < 9; ++e) { M.re[e] = W[base + 2*e]; M.im[e] = W[base + 2*e + 1]; }
  return M;
}
__device__ inline void storeW(float* W, int s, int ch, const C3& M) {
  int base = (s * NVAR_ + ch) * 18;
#pragma unroll
  for (int e = 0; e < 9; ++e) { W[base + 2*e] = M.re[e]; W[base + 2*e + 1] = M.im[e]; }
}

// ---------------- get_W for one site (plaquettes, polyakov, I, daggers) ----------
template <class LD>
__device__ void site_get_W(const LD& ld, int s, float* Wout) {
  int p = 0;
#pragma unroll
  for (int mu = 0; mu < 4; ++mu)
#pragma unroll
    for (int nu = mu + 1; nu < 4; ++nu) {
      C3 Um  = ld(s, mu);
      C3 Un1 = ld(shift_site(s, mu, 1), nu);
      C3 Um2 = ld(shift_site(s, nu, 1), mu);
      C3 Un  = ld(s, nu);
      C3 M = cmul(Um, Un1);
      M = cmul_dagB(M, Um2);
      M = cmul_dagB(M, Un);
      storeW(Wout, s, p, M);
      storeW(Wout, s, 11 + p, cdagger(M));
      ++p;
    }
#pragma unroll
  for (int mu = 0; mu < 4; ++mu) {
    C3 P = ld(s, mu);
    for (int k = 1; k < L_; ++k) P = cmul(P, ld(shift_site(s, mu, k), mu));
    storeW(Wout, s, 6 + mu, P);
    storeW(Wout, s, 17 + mu, cdagger(P));
  }
  C3 I = {};
  I.re[0] = I.re[4] = I.re[8] = 1.f;
  storeW(Wout, s, 10, I);
}

// ---------------- kernel 1: W = get_W(U) ----------------
__global__ void lcnn_w_from_u(const float* __restrict__ U_re,
                              const float* __restrict__ U_im,
                              float* __restrict__ Wbuf) {
  int s = blockIdx.x * blockDim.x + threadIdx.x;
  if (s >= NSITE) return;
  SplitLoader ld{U_re, U_im};
  site_get_W(ld, s, Wbuf);
}

// ---------------- kernel 3: out = get_W(EU) ----------------
__global__ void lcnn_w_from_eu(const float* __restrict__ EUbuf,
                               float* __restrict__ out) {
  int s = blockIdx.x * blockDim.x + threadIdx.x;
  if (s >= NSITE) return;
  InterLoader ld{EUbuf};
  site_get_W(ld, s, out);
}

// ---------------- closed-form exp(i*H), H = 3x3 traceless Hermitian ----------
__device__ inline void cscale_add(C3& D, float qr, float qi, const C3& X) {
  // D += (qr + i qi) * X
#pragma unroll
  for (int e = 0; e < 9; ++e) {
    D.re[e] = fmaf(qr, X.re[e], fmaf(-qi, X.im[e], D.re[e]));
    D.im[e] = fmaf(qr, X.im[e], fmaf( qi, X.re[e], D.im[e]));
  }
}

__device__ void expiH(const C3& H, C3& E) {
  C3 H2 = cmul(H, H);
  float c1 = 0.5f * (H2.re[0] + H2.re[4] + H2.re[8]);     // tr(H^2)/2, real >= 0
  if (c1 < 1e-12f) {                                       // exp(iH) ~ I + iH - H^2/2
#pragma unroll
    for (int e = 0; e < 9; ++e) {
      E.re[e] = -H.im[e] - 0.5f * H2.re[e];
      E.im[e] =  H.re[e] - 0.5f * H2.im[e];
    }
    E.re[0] += 1.f; E.re[4] += 1.f; E.re[8] += 1.f;
    return;
  }
  // det(H) (real for Hermitian)
  float m0r = H.re[4]*H.re[8]-H.im[4]*H.im[8] - (H.re[5]*H.re[7]-H.im[5]*H.im[7]);
  float m0i = H.re[4]*H.im[8]+H.im[4]*H.re[8] - (H.re[5]*H.im[7]+H.im[5]*H.re[7]);
  float m1r = H.re[3]*H.re[8]-H.im[3]*H.im[8] - (H.re[5]*H.re[6]-H.im[5]*H.im[6]);
  float m1i = H.re[3]*H.im[8]+H.im[3]*H.re[8] - (H.re[5]*H.im[6]+H.im[5]*H.re[6]);
  float m2r = H.re[3]*H.re[7]-H.im[3]*H.im[7] - (H.re[4]*H.re[6]-H.im[4]*H.im[6]);
  float m2i = H.re[3]*H.im[7]+H.im[3]*H.re[7] - (H.re[4]*H.im[6]+H.im[4]*H.re[6]);
  float c0 = (H.re[0]*m0r - H.im[0]*m0i)
           - (H.re[1]*m1r - H.im[1]*m1i)
           + (H.re[2]*m2r - H.im[2]*m2i);
  float sq  = sqrtf(c1 * (1.f/3.f));
  float arg = c0 / (2.f * sq * sq * sq);
  arg = fminf(1.f, fmaxf(-1.f, arg));
  float th = acosf(arg) * (1.f/3.f);
  const float W3 = 2.0943951023931953f; // 2*pi/3
  float l0 = 2.f*sq*cosf(th);
  float l1 = 2.f*sq*cosf(th - W3);
  float l2 = 2.f*sq*cosf(th + W3);
  float d0 = (l0-l1)*(l0-l2), d1 = (l1-l0)*(l1-l2), d2 = (l2-l0)*(l2-l1);
  d0 = (fabsf(d0) < 1e-6f) ? copysignf(1e-6f, d0) : d0;
  d1 = (fabsf(d1) < 1e-6f) ? copysignf(1e-6f, d1) : d1;
  d2 = (fabsf(d2) < 1e-6f) ? copysignf(1e-6f, d2) : d2;
  float e0r = cosf(l0), e0i = sinf(l0);
  float e1r = cosf(l1), e1i = sinf(l1);
  float e2r = cosf(l2), e2i = sinf(l2);
  // Lagrange interpolation: exp(iH) = q0 I + q1 H + q2 H^2
  float q0r = e0r*(l1*l2)/d0 + e1r*(l0*l2)/d1 + e2r*(l0*l1)/d2;
  float q0i = e0i*(l1*l2)/d0 + e1i*(l0*l2)/d1 + e2i*(l0*l1)/d2;
  float q1r = -e0r*(l1+l2)/d0 - e1r*(l0+l2)/d1 - e2r*(l0+l1)/d2;
  float q1i = -e0i*(l1+l2)/d0 - e1i*(l0+l2)/d1 - e2i*(l0+l1)/d2;
  float q2r = e0r/d0 + e1r/d1 + e2r/d2;
  float q2i = e0i/d0 + e1i/d1 + e2i/d2;
  C3 Z = {};
  Z.re[0] = Z.re[4] = Z.re[8] = 1.f;
  C3 R = {};
  cscale_add(R, q0r, q0i, Z);
  cscale_add(R, q1r, q1i, H);
  cscale_add(R, q2r, q2i, H2);
  E = R;
}

// ---------------- LDS fragment loads matching gfx1250 16-bit WMMA layouts ------
__device__ inline v16h load_a_frag(const _Float16* A, int rowBase, int lane, int kk) {
  // 16-bit A 16x32: lanes 0-15 -> K {0..7,16..23}; lanes 16-31 -> K {8..15,24..31}
  int row = rowBase + (lane & 15);
  int half = (lane < 16) ? 0 : 8;
  const _Float16* p = A + row * ASTR + kk * 32;
  v16h r;
#pragma unroll
  for (int i = 0; i < 8; ++i) { r[i] = p[half + i]; r[8 + i] = p[16 + half + i]; }
  return r;
}
__device__ inline v16h load_b_frag(const _Float16* Bt, int lane, int kk) {
  // 16-bit B 32x16: lanes 0-15 -> N=lane, K {0..7,16..23} ; lanes 16-31 -> N=lane-16, K {8..15,24..31}
  int n = lane & 15;
  int kb = kk * 32 + ((lane < 16) ? 0 : 8);
  const _Float16* p = Bt + n * ASTR + kb;
  v16h r;
#pragma unroll
  for (int i = 0; i < 8; ++i) { r[i] = p[i]; r[8 + i] = p[16 + i]; }
  return r;
}

// ---------------- kernel 2: Wp -> (WMMA) Wc -> Wb -> exp -> EU ----------------
__global__ __launch_bounds__(WGT)
void lcnn_main(const float* __restrict__ U_re, const float* __restrict__ U_im,
               const float* __restrict__ Wbuf,
               const float* __restrict__ omega,   // [8][21][4][3] = [8][252]
               const float* __restrict__ alpha,   // [8][8][8]
               const float* __restrict__ beta,    // [4][8]
               float* __restrict__ EUbuf) {       // interleaved [site][mu][9][2]
  extern __shared__ __align__(16) char smem[];
  _Float16* Als = (_Float16*)smem;                    // [MROWS][ASTR]
  _Float16* Bls = Als + MROWS * ASTR;                 // [16][ASTR]
  float*    Wc  = (float*)(Bls + 16 * ASTR);          // [SPW][8][9][2]

  const int tid  = threadIdx.x;
  const int wave = tid >> 5;
  const int lane = tid & 31;
  const int s0   = blockIdx.x * SPW;

  __builtin_prefetch(omega, 0, 1);  // global_prefetch_b8: warm the weights

  // ---- stage 0: B tile = omega^T (padded), zero A pad columns ----
  for (int idx = tid; idx < 16 * KPAD; idx += WGT) {
    int n = idx / KPAD, k = idx % KPAD;
    Bls[n * ASTR + k] = (n < NOUT_ && k < KDIM) ? (_Float16)omega[n * KDIM + k]
                                                : (_Float16)0.f;
  }
  for (int idx = tid; idx < MROWS * (KPAD - KDIM); idx += WGT) {
    int row = idx / (KPAD - KDIM), k = KDIM + idx % (KPAD - KDIM);
    Als[row * ASTR + k] = (_Float16)0.f;
  }

  // ---- stage 1: Wp[site, i, m, k] = T W_shift T^dagger  ->  f16 A tile ----
  for (int idx = tid; idx < SPW * KDIM; idx += WGT) {
    int sl = idx / KDIM, col = idx % KDIM;
    int iv = col / 12, rem = col % 12, m = rem / 3, kk = rem % 3; // k in {-1,0,1}
    int s  = s0 + sl;
    C3 Wm = loadW(Wbuf, shift_site(s, m, kk - 1), iv);
    C3 P;
    if (kk == 1) {                        // T = I
      P = Wm;
    } else if (kk == 2) {                 // T = U_mu(x)
      SplitLoader ld{U_re, U_im};
      C3 T = ld(s, m);
      P = cmul(T, Wm); P = cmul_dagB(P, T);
    } else {                              // T = U_mu(x - mu)^dagger
      SplitLoader ld{U_re, U_im};
      C3 T = cdagger(ld(shift_site(s, m, -1), m));
      P = cmul(T, Wm); P = cmul_dagB(P, T);
    }
    int rb = sl * 18;
#pragma unroll
    for (int e = 0; e < 9; ++e) {
      Als[(rb + 2*e    ) * ASTR + col] = (_Float16)P.re[e];
      Als[(rb + 2*e + 1) * ASTR + col] = (_Float16)P.im[e];
    }
  }
  __syncthreads();

  // ---- stage 2: Wc = omega . Wp via v_wmma_f32_16x16x32_f16 ----
  v16h bfrag[KSTEPS];
#pragma unroll
  for (int kk = 0; kk < KSTEPS; ++kk) bfrag[kk] = load_b_frag(Bls, lane, kk);

  for (int t = wave; t < MROWS / 16; t += WGT / 32) {   // uniform per wave
    // Issue ALL fragment loads first so the WMMA chain overlaps LDS latency
    // (partial s_wait_dscnt instead of a full wait before every WMMA).
    v16h afrag[KSTEPS];
#pragma unroll
    for (int kk = 0; kk < KSTEPS; ++kk)
      afrag[kk] = load_a_frag(Als, t * 16, lane, kk);

    v8f acc = {};
#pragma unroll
    for (int kk = 0; kk < KSTEPS; ++kk)
      acc = __builtin_amdgcn_wmma_f32_16x16x32_f16(
          false, afrag[kk], false, bfrag[kk], (short)0, acc, false, false);

    int n = lane & 15;
    if (n < NOUT_) {
      // 32-bit C/D layout: lane = N column; VGPR r -> row r (lanes 0-15) / 8+r.
      int row0 = t * 16 + ((lane < 16) ? 0 : 8);
      int sl = row0 / 18, comp = row0 % 18;      // divide once, then increment
#pragma unroll
      for (int r = 0; r < 8; ++r) {
        Wc[((sl * NOUT_ + n) * 9 + (comp >> 1)) * 2 + (comp & 1)] = acc[r];
        if (++comp == 18) { comp = 0; ++sl; }
      }
    }
  }
  __syncthreads();

  // ---- stage 3: per-site bilinear + projection + exp(iH) + EU ----
  if (tid < SPW) {
    int s = s0 + tid;
    C3 wc[NOUT_];
#pragma unroll
    for (int o = 0; o < NOUT_; ++o)
#pragma unroll
      for (int e = 0; e < 9; ++e) {
        wc[o].re[e] = Wc[((tid * NOUT_ + o) * 9 + e) * 2 + 0];
        wc[o].im[e] = Wc[((tid * NOUT_ + o) * 9 + e) * 2 + 1];
      }

    C3 Wb[NCH_] = {};
    for (int j = 0; j < NOUT_; ++j)
      for (int k = 0; k < NOUT_; ++k) {
        C3 M = cmul(wc[j], wc[k]);
        for (int c = 0; c < NCH_; ++c) {
          float al = alpha[(c * NOUT_ + j) * NOUT_ + k];
#pragma unroll
          for (int e = 0; e < 9; ++e) {
            Wb[c].re[e] = fmaf(al, M.re[e], Wb[c].re[e]);
            Wb[c].im[e] = fmaf(al, M.im[e], Wb[c].im[e]);
          }
        }
      }

    // f * Wb, anti-hermitian traceless (in place)
    for (int c = 0; c < NCH_; ++c) {
      float f = Wb[c].re[0] + Wb[c].re[4] + Wb[c].re[8];
      C3 Wa;
#pragma unroll
      for (int e = 0; e < 9; ++e) { Wa.re[e] = f * Wb[c].re[e]; Wa.im[e] = f * Wb[c].im[e]; }
      C3 Ah;
#pragma unroll
      for (int a = 0; a < 3; ++a)
#pragma unroll
        for (int b = 0; b < 3; ++b) {
          Ah.re[a*3+b] = 0.5f * (Wa.re[a*3+b] - Wa.re[b*3+a]);
          Ah.im[a*3+b] = 0.5f * (Wa.im[a*3+b] + Wa.im[b*3+a]);
        }
      float ti = (Ah.im[0] + Ah.im[4] + Ah.im[8]) * (1.f/3.f); // trace purely imaginary
      Ah.im[0] -= ti; Ah.im[4] -= ti; Ah.im[8] -= ti;
      Wb[c] = Ah;
    }

    SplitLoader ld{U_re, U_im};
    for (int m = 0; m < D_; ++m) {
      // H = sum_c beta[m,c] * (i * Wah_c)   (Hermitian traceless)
      C3 H = {};
      for (int c = 0; c < NCH_; ++c) {
        float bt = beta[m * NCH_ + c];
#pragma unroll
        for (int e = 0; e < 9; ++e) {
          H.re[e] = fmaf(bt, -Wb[c].im[e], H.re[e]);
          H.im[e] = fmaf(bt,  Wb[c].re[e], H.im[e]);
        }
      }
      C3 E;
      expiH(H, E);
      C3 EU = cmul(E, ld(s, m));
      int base = (s * 4 + m) * 18;
#pragma unroll
      for (int e = 0; e < 9; ++e) {
        EUbuf[base + 2*e]     = EU.re[e];
        EUbuf[base + 2*e + 1] = EU.im[e];
      }
    }
  }
}

// ---------------- host-side launch ----------------
extern "C" void kernel_launch(void* const* d_in, const int* in_sizes, int n_in,
                              void* d_out, int out_size, void* d_ws, size_t ws_size,
                              hipStream_t stream) {
  const float* U_re  = (const float*)d_in[0];
  const float* U_im  = (const float*)d_in[1];
  const float* omega = (const float*)d_in[2];
  const float* alpha = (const float*)d_in[3];
  const float* beta  = (const float*)d_in[4];
  float* out = (float*)d_out;

  float* Wbuf  = (float*)d_ws;                       // NSITE*21*18 floats (~12.4 MB)
  float* EUbuf = Wbuf + (size_t)NSITE * NVAR_ * 18;  // NSITE*4*18 floats  (~2.4 MB)

  lcnn_w_from_u<<<NSITE / 128, 128, 0, stream>>>(U_re, U_im, Wbuf);

  size_t lds = (size_t)MROWS * ASTR * 2      // A tile (f16)
             + (size_t)16 * ASTR * 2         // B tile (f16)
             + (size_t)SPW * NOUT_ * 18 * 4; // Wc (f32)
  lcnn_main<<<NSITE / SPW, WGT, lds, stream>>>(U_re, U_im, Wbuf,
                                               omega, alpha, beta, EUbuf);

  lcnn_w_from_eu<<<NSITE / 128, 128, 0, stream>>>(EUbuf, out);
}
